// AdaptiveInputEmbedding_81174881894966
// MI455X (gfx1250) — matrix operand order repaired and checked by reading, not compiled, not run on hardware
//
#include <hip/hip_runtime.h>

// ---------- CDNA5 wave32 WMMA types ----------
typedef __attribute__((ext_vector_type(16))) __bf16    v16bf;
typedef __attribute__((ext_vector_type(8)))  float     v8f;
typedef __attribute__((ext_vector_type(4)))  float     v4f;
typedef __attribute__((ext_vector_type(8)))  unsigned  v8u;

// round-to-nearest-even f32 -> bf16, result in bits [31:16]
__device__ __forceinline__ unsigned rne_hi(float f) {
  unsigned u = __builtin_bit_cast(unsigned, f);
  return u + 0x7FFFu + ((u >> 16) & 1u);
}

// pack two f32 -> {hi_bf16, lo_bf16} in one dword
__device__ __forceinline__ unsigned f2bf_pk(float lo, float hi) {
  // v_perm_b32: result bytes {hi[3],hi[2],lo[3],lo[2]}
  return __builtin_amdgcn_perm(rne_hi(hi), rne_hi(lo), 0x07060302u);
}

// One bucket: masked A(16 x d) @ W(d x 128) accumulated into 8 16x16 f32 tiles.
// B is staged in double-buffered LDS (f32->bf16, pre-swizzled to the ISA
// B-fragment layout) and shared by all 8 waves of the block: one barrier per
// K-chunk. All threads run the staging loop and the barrier; WMMA/A-load work
// is skipped by waves with no token in the bucket.
__device__ __forceinline__ void process_bucket(
    int d, const float* __restrict__ emb, const float* __restrict__ W,
    int lo, int size, int token,
    int tid, int lane, int khalf, int colBase,
    bool blockAct, bool waveAct,
    __bf16* ldsB, int& phase, v8f acc[8]) {
  if (!blockAct) return;  // uniform over block (shared-mem flag) -> barrier-safe

  const bool inb = (token >= lo) && (token < lo + size);
  int idx = token - lo;
  idx = idx < 0 ? 0 : idx;
  idx = idx >= size ? size - 1 : idx;          // clamped safe gather (as reference)
  const float* row = emb + (size_t)idx * d + khalf * 8;
  const float msk = inb ? 1.0f : 0.0f;         // zero A rows for out-of-bucket tokens

#pragma unroll 1
  for (int kc = 0; kc < d; kc += 32) {
    __bf16*   buf  = ldsB + (phase & 1) * 4096;  // 8 KB ping-pong buffer
    unsigned* bufu = (unsigned*)buf;
    phase ^= 1;

    // --- A fragment (private, gathered). ISA 16-bit A 16x32 layout:
    //     lane l (group g=l>>4): e<8 -> K = 8g+e ; e>=8 -> K = 16+8g+(e-8)
    v16bf a = {};
    if (waveAct) {
      const float* p = row + kc;
      if (kc + 32 < d) __builtin_prefetch(p + 32, 0, 1);   // global_prefetch_b8
      v4f f0 = *(const v4f*)(p + 0);    // e = 0..3
      v4f f1 = *(const v4f*)(p + 4);    // e = 4..7
      v4f f2 = *(const v4f*)(p + 16);   // e = 8..11
      v4f f3 = *(const v4f*)(p + 20);   // e = 12..15
      f0 *= msk; f1 *= msk; f2 *= msk; f3 *= msk;
      v8u ap;
      ap[0] = f2bf_pk(f0[0], f0[1]);
      ap[1] = f2bf_pk(f0[2], f0[3]);
      ap[2] = f2bf_pk(f1[0], f1[1]);
      ap[3] = f2bf_pk(f1[2], f1[3]);
      ap[4] = f2bf_pk(f2[0], f2[1]);
      ap[5] = f2bf_pk(f2[2], f2[3]);
      ap[6] = f2bf_pk(f3[0], f3[1]);
      ap[7] = f2bf_pk(f3[2], f3[3]);
      a = __builtin_bit_cast(v16bf, ap);
    }

    // --- stage B into the ping-pong buffer: each thread converts two adjacent
    //     K rows (same column) of W[kc..kc+32, colBase..colBase+128] and writes
    //     one packed dword to LDS in B-fragment order: lane 16g+n, e<8 -> K=8g+e ; e>=8 -> K=16+8g+(e-8).
    //     Safe without a pre-barrier: this buffer's previous readers (two
    //     chunks ago) drained their ds_loads before passing the last barrier.
    for (int j = tid; j < 2048; j += 256) {
      const int kp = j >> 7;         // K-pair index 0..15
      const int c  = j & 127;        // column 0..127 (coalesced)
      const int k0 = kp * 2;
      const float* wp = W + (size_t)(kc + k0) * 1024 + colBase + c;
      const float wv0 = wp[0];
      const float wv1 = wp[1024];
      const int t    = c >> 4;       // N-tile 0..7
      const int nloc = c & 15;
      const int g    = (k0 >> 3) & 1;      // lane group
      const int e0   = (k0 & 7) | ((k0 >> 4) << 3);      // even element index
      bufu[((t * 32) + (g * 16 + nloc)) * 8 + (e0 >> 1)] = f2bf_pk(wv0, wv1);
    }
    __syncthreads();  // staged B visible (single barrier per chunk)

    if (waveAct) {
      const v16bf* bp = (const v16bf*)buf;
      // Load ALL 8 B fragments first (16 ds_load_b128 in flight), then run
      // the 8 WMMAs back-to-back: no per-WMMA full dscnt drain.
      v16bf bf[8];
#pragma unroll
      for (int t = 0; t < 8; ++t) bf[t] = bp[t * 32 + lane];
#pragma unroll
      for (int t = 0; t < 8; ++t) {
        acc[t] = __builtin_amdgcn_wmma_f32_16x16x32_bf16(
            false, a, false, bf[t], (short)0, acc[t], false, false);
      }
    }
  }
}

__global__ void __launch_bounds__(256)
adaptive_embed_kernel(const int* __restrict__ x,
                      const float* __restrict__ e0, const float* __restrict__ e1,
                      const float* __restrict__ e2, const float* __restrict__ e3,
                      const float* __restrict__ w0, const float* __restrict__ w1,
                      const float* __restrict__ w2, const float* __restrict__ w3,
                      float* __restrict__ out) {
  __shared__ __attribute__((aligned(32))) __bf16 ldsB[2 * 8 * 32 * 16];  // 16 KB
  __shared__ int anyWB[8 * 4];  // [wave][bucket] any-token flags
  __shared__ int anyB[4];       // [bucket] block-level flags

  const int tid   = threadIdx.x;
  const int wave  = tid >> 5;
  const int lane  = tid & 31;
  const int mrow  = lane & 15;
  const int khalf = lane >> 4;
  const int tokBlock = blockIdx.x * 128;  // 128 tokens per block
  const int colBase  = blockIdx.y * 128;  // 128 output cols per block

  if (tid < 32) { anyWB[tid] = 0; if (tid < 4) anyB[tid] = 0; }
  __syncthreads();
  if (tid < 128) {
    const int t = x[tokBlock + tid];
    const int b = (t < 300) ? 0 : (t < 3000) ? 1 : (t < 30000) ? 2 : 3;
    anyWB[(tid >> 4) * 4 + b] = 1;  // benign same-value races
    anyB[b] = 1;
  }
  const int token = x[tokBlock + wave * 16 + mrow];
  __syncthreads();

  const bool bA0 = anyB[0] != 0, bA1 = anyB[1] != 0;
  const bool bA2 = anyB[2] != 0, bA3 = anyB[3] != 0;
  const bool wA0 = __builtin_amdgcn_readfirstlane(anyWB[wave * 4 + 0]) != 0;
  const bool wA1 = __builtin_amdgcn_readfirstlane(anyWB[wave * 4 + 1]) != 0;
  const bool wA2 = __builtin_amdgcn_readfirstlane(anyWB[wave * 4 + 2]) != 0;
  const bool wA3 = __builtin_amdgcn_readfirstlane(anyWB[wave * 4 + 3]) != 0;

  v8f acc[8] = {};
  int phase = 0;  // uniform across block (bucket skip is block-uniform)

  process_bucket(1024, e0, w0,     0,    300, token, tid, lane, khalf, colBase, bA0, wA0, ldsB, phase, acc);
  process_bucket( 512, e1, w1,   300,   2700, token, tid, lane, khalf, colBase, bA1, wA1, ldsB, phase, acc);
  process_bucket( 256, e2, w2,  3000,  27000, token, tid, lane, khalf, colBase, bA2, wA2, ldsB, phase, acc);
  process_bucket( 128, e3, w3, 30000, 237734, token, tid, lane, khalf, colBase, bA3, wA3, ldsB, phase, acc);

  // D tile layout: VGPR r -> M = r + 8*khalf ; N = lane & 15
  const int rowBase = tokBlock + wave * 16 + khalf * 8;
#pragma unroll
  for (int t = 0; t < 8; ++t) {
    const int col = colBase + t * 16 + mrow;
#pragma unroll
    for (int r = 0; r < 8; ++r) {
      out[(size_t)(rowBase + r) * 1024 + col] = acc[t][r];
    }
  }
}

extern "C" void kernel_launch(void* const* d_in, const int* in_sizes, int n_in,
                              void* d_out, int out_size, void* d_ws, size_t ws_size,
                              hipStream_t stream) {
  (void)in_sizes; (void)n_in; (void)out_size; (void)d_ws; (void)ws_size;
  const int*   x  = (const int*)d_in[0];
  const float* e0 = (const float*)d_in[1];
  const float* e1 = (const float*)d_in[2];
  const float* e2 = (const float*)d_in[3];
  const float* e3 = (const float*)d_in[4];
  const float* w0 = (const float*)d_in[5];
  const float* w1 = (const float*)d_in[6];
  const float* w2 = (const float*)d_in[7];
  const float* w3 = (const float*)d_in[8];
  float* out = (float*)d_out;

  dim3 grid(32768 / 128, 1024 / 128, 1);  // 256 token groups x 8 col groups
  adaptive_embed_kernel<<<grid, 256, 0, stream>>>(x, e0, e1, e2, e3,
                                                  w0, w1, w2, w3, out);
}
